// SpatialAttentionMultiScale_4226247819455
// MI455X (gfx1250) — matrix operand order repaired and checked by reading, not compiled, not run on hardware
//
#include <hip/hip_runtime.h>
#include <hip/hip_fp16.h>

typedef __attribute__((ext_vector_type(16))) _Float16 v16h;
typedef __attribute__((ext_vector_type(8)))  float    v8f;

#define B_   16
#define C_   256
#define H_   128
#define W_   128
#define HW_  (H_*W_)

// ---------------------------------------------------------------------------
// Pass 1: channel reduction. comb[b][0][p] = mean_c x, comb[b][1][p] = max_c x
// Pure bandwidth: 256 MB coalesced read, 2 MB write. float2 per thread so the
// grid is 512 workgroups (4096 wave32s) — enough to fill every WGP while each
// wave load still covers two full cachelines (32 lanes * 8 B).
// grid = B_*32 blocks, 256 threads; each block owns 512 pixels of one batch.
// ---------------------------------------------------------------------------
__global__ __launch_bounds__(256) void sam_reduce(const float* __restrict__ x,
                                                  float* __restrict__ comb) {
  int b     = blockIdx.x >> 5;
  int chunk = blockIdx.x & 31;
  int p2    = chunk * 256 + threadIdx.x;                 // float2 idx in plane [0,8192)
  const float2* xp = reinterpret_cast<const float2*>(x + (size_t)b * C_ * HW_) + p2;
  float2 s = make_float2(0.f, 0.f);
  float2 m = make_float2(-3.4e38f, -3.4e38f);
  #pragma unroll 8
  for (int c = 0; c < C_; ++c) {
    float2 v = xp[(size_t)c * (HW_ / 2)];
    s.x += v.x; s.y += v.y;
    m.x = fmaxf(m.x, v.x); m.y = fmaxf(m.y, v.y);
  }
  const float inv = 1.0f / (float)C_;
  float2 a = make_float2(s.x * inv, s.y * inv);
  reinterpret_cast<float2*>(comb + ((size_t)b * 2 + 0) * HW_)[p2] = a;
  reinterpret_cast<float2*>(comb + ((size_t)b * 2 + 1) * HW_)[p2] = m;
}

// ---------------------------------------------------------------------------
// Pass 2: fused 7x7 conv (w7 + center-embedded w5 + w3) + sigmoid, via WMMA.
// Implicit GEMM: D[m,n] = sum_k A[m,k]*B[k,n], M = 16 pixels along a row,
// K = 98 im2col features (2 ch * 7 * 7) padded to 128, B columns replicated
// with the fused weights so every output column holds the conv result.
// One block per (batch, row): 8 waves x 16 pixels = 128-wide row.
// ---------------------------------------------------------------------------
__global__ __launch_bounds__(256) void sam_conv_wmma(const float* __restrict__ comb,
                                                     const float* __restrict__ w3,
                                                     const float* __restrict__ w5,
                                                     const float* __restrict__ w7,
                                                     float* __restrict__ att) {
  __shared__ float tile[2][7][144];   // rows y-3..y+3, cols -3..130 (+pad), 2 channels
  __shared__ float wlds[128];         // fused 7x7x2 weights, k = c*49 + ky*7 + kx

  int b = blockIdx.x >> 7;
  int y = blockIdx.x & 127;

  for (int i = threadIdx.x; i < 2 * 7 * 144; i += 256) {
    int c   = i / (7 * 144);
    int rem = i % (7 * 144);
    int yy  = rem / 144;
    int xx  = rem % 144;
    int gy  = y + yy - 3;
    int gx  = xx - 3;
    float v = 0.f;
    if (gy >= 0 && gy < H_ && gx >= 0 && gx < W_)
      v = comb[((size_t)b * 2 + c) * HW_ + gy * W_ + gx];
    tile[c][yy][xx] = v;
  }
  if (threadIdx.x < 128) {
    int k = threadIdx.x;
    float wv = 0.f;
    if (k < 98) {
      int c = k / 49, r = k % 49, ky = r / 7, kx = r % 7;
      wv = w7[c * 49 + ky * 7 + kx];
      if (ky >= 1 && ky <= 5 && kx >= 1 && kx <= 5) wv += w5[c * 25 + (ky - 1) * 5 + (kx - 1)];
      if (ky >= 2 && ky <= 4 && kx >= 2 && kx <= 4) wv += w3[c * 9 + (ky - 2) * 3 + (kx - 2)];
    }
    wlds[k] = wv;
  }
  __syncthreads();

  int wave = threadIdx.x >> 5;
  int lane = threadIdx.x & 31;
  int m    = lane & 15;     // output pixel row-within-tile (A matrix M index)
  int half = lane >> 4;     // lane group selects which K sub-block this lane holds
  int x0   = wave * 16;     // first output pixel of this wave

  v8f acc = {0.f, 0.f, 0.f, 0.f, 0.f, 0.f, 0.f, 0.f};
  #pragma unroll
  for (int chunk = 0; chunk < 4; ++chunk) {
    v16h a, bm;
    #pragma unroll
    for (int e = 0; e < 16; ++e) {
      // A 16x32 f16 layout (ISA 7.12.2): lanes 0-15 hold K {0..7,16..23},
      // lanes 16-31 hold K {8..15,24..31} for row M = lane%16.
      int ka = chunk * 32 + ((e < 8) ? (half * 8 + e) : (16 + half * 8 + (e - 8)));
      float av = 0.f;
      if (ka < 98) {
        int c = ka / 49, r = ka % 49, dy = r / 7, dx = r % 7;
        av = tile[c][dy][x0 + m + dx];
      }
      a[e] = (_Float16)av;
      // B 32x16 f16 layout: lane holds column N = lane%16, same K order as A.
      // Columns replicated -> D identical across N.
      int kb = chunk * 32 + ((e < 8) ? (half * 8 + e) : (16 + half * 8 + (e - 8)));
      bm[e] = (_Float16)((kb < 98) ? wlds[kb] : 0.f);
    }
    acc = __builtin_amdgcn_wmma_f32_16x16x32_f16(false, a, false, bm,
                                                 (short)0, acc, false, false);
  }

  // D layout: VGPR r, lane l -> M = r + 8*(l/16). Lanes 0 and 16 cover M 0..15.
  if ((lane & 15) == 0) {
    float* dst = att + (size_t)b * HW_ + y * W_ + x0 + half * 8;
    #pragma unroll
    for (int r = 0; r < 8; ++r) {
      float f = acc[r];
      dst[r] = 1.0f / (1.0f + __expf(-f));
    }
  }
}

// ---------------------------------------------------------------------------
// Pass 3: out = x * attention (broadcast over channels).
// Pure bandwidth: 256 MB read + 256 MB write, float4 coalesced. The 1 MB
// attention map is hot in L2 across all 256 channel re-reads.
// ---------------------------------------------------------------------------
__global__ __launch_bounds__(256) void sam_scale(const float* __restrict__ x,
                                                 const float* __restrict__ att,
                                                 float* __restrict__ out) {
  int idx = blockIdx.x * 256 + threadIdx.x;   // float4 index, [0, 2^24)
  const float4* x4 = reinterpret_cast<const float4*>(x);
  const float4* a4 = reinterpret_cast<const float4*>(att);
  float4*       o4 = reinterpret_cast<float4*>(out);
  int b  = idx >> 20;       // C_*HW_/4 = 2^20 float4 per batch
  int p4 = idx & 4095;      // HW_/4 = 4096 float4 per plane
  float4 xv = x4[idx];
  float4 av = a4[(b << 12) + p4];
  o4[idx] = make_float4(xv.x * av.x, xv.y * av.y, xv.z * av.z, xv.w * av.w);
}

// ---------------------------------------------------------------------------
extern "C" void kernel_launch(void* const* d_in, const int* in_sizes, int n_in,
                              void* d_out, int out_size, void* d_ws, size_t ws_size,
                              hipStream_t stream) {
  const float* x  = (const float*)d_in[0];
  const float* w3 = (const float*)d_in[1];
  const float* w5 = (const float*)d_in[2];
  const float* w7 = (const float*)d_in[3];
  float* out = (float*)d_out;

  float* comb = (float*)d_ws;                       // B_*2*HW_ floats (2 MB)
  float* att  = comb + (size_t)B_ * 2 * HW_;        // B_*HW_ floats (1 MB)

  sam_reduce   <<<B_ * 32, 256, 0, stream>>>(x, comb);
  sam_conv_wmma<<<B_ * H_, 256, 0, stream>>>(comb, w3, w5, w7, att);
  sam_scale    <<<(B_ * C_ * HW_ / 4) / 256, 256, 0, stream>>>(x, att, out);
}
